// Model_75857712382035
// MI455X (gfx1250) — matrix operand order repaired and checked by reading, not compiled, not run on hardware
//
#include <hip/hip_runtime.h>
#include <hip/hip_bf16.h>

typedef __attribute__((ext_vector_type(16))) _Float16 v16h;
typedef __attribute__((ext_vector_type(8)))  _Float16 v8h;
typedef __attribute__((ext_vector_type(8)))  float    v8f;

// Problem dims
#define Bsz  16384
#define Fdim 1024
#define Eexp 8
#define Udim 512
#define EH1c 1024
#define EH2c 1024
#define TH1c 512
#define TH2c 256
#define Tt   4

#define BM 128
#define BN 128
#define BK 32
#define LDSK (BK + 8)   // 40 halfwords = 80B row stride (16B aligned)

union Frag16 { v16h v; v8h h[2]; };

// ---------------------------------------------------------------------------
// fp32 -> f16 conversion (grid-stride)
// ---------------------------------------------------------------------------
__global__ void cvt_f32_f16(const float* __restrict__ s, _Float16* __restrict__ d,
                            long long n) {
    long long i = (long long)blockIdx.x * blockDim.x + threadIdx.x;
    long long stride = (long long)gridDim.x * blockDim.x;
    for (; i < n; i += stride) d[i] = (_Float16)s[i];
}

// ---------------------------------------------------------------------------
// Batched f16 WMMA GEMM:  D[bz] = act(A[bz] @ W[bz] + bias[bz])
//   A: [M,K] row-major f16, W: [K,N] row-major f16, bias: [N] f32, D: [M,N] f16
//   Block tile 128x128, 8 waves of 32 lanes, each wave -> 32x64 (2x4 frags)
// ---------------------------------------------------------------------------
__global__ __launch_bounds__(256) void wmma_gemm_f16(
    const _Float16* __restrict__ A, long long strideA, int lda,
    const _Float16* __restrict__ W, long long strideW, int ldw,
    const float*    __restrict__ bias, int strideBias,
    _Float16*       __restrict__ D, long long strideD, int ldd,
    int M, int N, int K, int doRelu)
{
    __shared__ _Float16 As[BM][LDSK];
    __shared__ _Float16 Bs[BN][LDSK];   // transposed: [n][k]

    const int bz = blockIdx.z;
    A    += (long long)bz * strideA;
    W    += (long long)bz * strideW;
    bias += (long long)bz * strideBias;
    D    += (long long)bz * strideD;

    const int tid  = threadIdx.x;
    const int lane = tid & 31;
    const int wid  = tid >> 5;     // 0..7
    const int wm   = wid >> 1;     // 0..3  (M direction, 32 rows each)
    const int wn   = wid & 1;      // 0..1  (N direction, 64 cols each)

    const int m0 = blockIdx.y * BM;
    const int n0 = blockIdx.x * BN;

    v8f acc[2][4];
#pragma unroll
    for (int i = 0; i < 2; ++i)
#pragma unroll
        for (int j = 0; j < 4; ++j)
            acc[i][j] = (v8f){0.f,0.f,0.f,0.f,0.f,0.f,0.f,0.f};

    // A-tile loader: each thread moves 16 f16 (two 16B chunks)
    const int arow  = tid >> 1;
    const int akseg = (tid & 1) * 16;
    // B-tile loader: each thread gathers a 16-deep K column for one N
    const int bcol = tid & (BN - 1);
    const int bkh  = tid >> 7;     // 0 or 1

    for (int k0 = 0; k0 < K; k0 += BK) {
        {   // global -> LDS, A (coalesced 16B vector loads)
            const v8h* gp = (const v8h*)(A + (long long)(m0 + arow) * lda + k0 + akseg);
            v8h a0 = gp[0];
            v8h a1 = gp[1];
            *(v8h*)&As[arow][akseg]     = a0;
            *(v8h*)&As[arow][akseg + 8] = a1;
        }
        {   // global -> LDS, W transposed into Bs[n][k]
            v8h b0, b1;
#pragma unroll
            for (int kk = 0; kk < 8; ++kk) {
                b0[kk] = W[(long long)(k0 + bkh*16 + kk)     * ldw + n0 + bcol];
                b1[kk] = W[(long long)(k0 + bkh*16 + kk + 8) * ldw + n0 + bcol];
            }
            *(v8h*)&Bs[bcol][bkh*16]     = b0;
            *(v8h*)&Bs[bcol][bkh*16 + 8] = b1;
        }
        __syncthreads();

        // A fragments: 16x32, lanes 0-15 rows M with K {0..7,16..23},
        // lanes 16-31 same rows with K {8..15,24..31}
        Frag16 af[2];
#pragma unroll
        for (int mf = 0; mf < 2; ++mf) {
            const int row  = wm*32 + mf*16 + (lane & 15);
            const int koff = (lane >> 4) * 8;
            af[mf].h[0] = *(const v8h*)&As[row][koff];
            af[mf].h[1] = *(const v8h*)&As[row][koff + 16];
        }
        // B fragments: 32x16, lane = column, lanes 0-15 K {0..7,16..23}, lanes 16-31 K {8..15,24..31}
        Frag16 bf[4];
#pragma unroll
        for (int nf = 0; nf < 4; ++nf) {
            const int col = wn*64 + nf*16 + (lane & 15);
            const int kst = (lane >> 4) * 8;
            bf[nf].h[0] = *(const v8h*)&Bs[col][kst];
            bf[nf].h[1] = *(const v8h*)&Bs[col][kst + 16];
        }

#pragma unroll
        for (int mf = 0; mf < 2; ++mf)
#pragma unroll
            for (int nf = 0; nf < 4; ++nf)
                acc[mf][nf] = __builtin_amdgcn_wmma_f32_16x16x32_f16(
                    false, af[mf].v, false, bf[nf].v,
                    (short)0, acc[mf][nf], false, false);

        __syncthreads();
    }

    // Epilogue: C/D layout — lane&15 = N, lane>>4 selects M+8, vgpr r = M row
#pragma unroll
    for (int mf = 0; mf < 2; ++mf) {
#pragma unroll
        for (int nf = 0; nf < 4; ++nf) {
            const int col = n0 + wn*64 + nf*16 + (lane & 15);
            const float bv = bias[col];
            const int rowbase = m0 + wm*32 + mf*16 + (lane >> 4) * 8;
#pragma unroll
            for (int r = 0; r < 8; ++r) {
                float v = acc[mf][nf][r] + bv;
                if (doRelu) v = v > 0.f ? v : 0.f;
                D[(long long)(rowbase + r) * ldd + col] = (_Float16)v;
            }
        }
    }
}

// ---------------------------------------------------------------------------
// Per-task gates: logits[t,b,e] = x[b,:] @ Wg[t,:,e]; softmax over E=8
// grid (B/256, T); Wg[t] cached in LDS (32 KB)
// ---------------------------------------------------------------------------
__global__ __launch_bounds__(256) void gates_kernel(
    const float* __restrict__ x, const float* __restrict__ Wg,
    float* __restrict__ gates)
{
    __shared__ float WgS[Fdim][Eexp];
    const int t = blockIdx.y;
    const float* Wgt = Wg + (long long)t * Fdim * Eexp;
    for (int i = threadIdx.x; i < Fdim * Eexp; i += 256)
        ((float*)WgS)[i] = Wgt[i];
    __syncthreads();

    const int b = blockIdx.x * 256 + threadIdx.x;
    const float* xb = x + (long long)b * Fdim;
    float acc[Eexp] = {};
    for (int f = 0; f < Fdim; ++f) {
        const float xv = xb[f];
#pragma unroll
        for (int e = 0; e < Eexp; ++e) acc[e] += xv * WgS[f][e];
    }
    float m = acc[0];
#pragma unroll
    for (int e = 1; e < Eexp; ++e) m = acc[e] > m ? acc[e] : m;
    float s = 0.f;
#pragma unroll
    for (int e = 0; e < Eexp; ++e) { acc[e] = __expf(acc[e] - m); s += acc[e]; }
    const float inv = 1.f / s;
    float* gp = gates + ((long long)t * Bsz + b) * Eexp;
#pragma unroll
    for (int e = 0; e < Eexp; ++e) gp[e] = acc[e] * inv;
}

// ---------------------------------------------------------------------------
// Combine: ti[t,b,u] = sum_e gates[t,b,e] * eout[e,b,u]   (f16 out)
// ---------------------------------------------------------------------------
__global__ __launch_bounds__(256) void combine_kernel(
    const float* __restrict__ gates, const _Float16* __restrict__ eout,
    _Float16* __restrict__ ti)
{
    const long long idx = (long long)blockIdx.x * 256 + threadIdx.x; // t*B*U + b*U + u
    const long long u  = idx % Udim;
    const long long tb = idx / Udim;
    const long long b  = tb % Bsz;
    const long long t  = tb / Bsz;
    const float* g = gates + (t * Bsz + b) * Eexp;
    float acc = 0.f;
#pragma unroll
    for (int e = 0; e < Eexp; ++e)
        acc += g[e] * (float)eout[((long long)e * Bsz + b) * Udim + u];
    ti[idx] = (_Float16)acc;
}

// ---------------------------------------------------------------------------
// Tower final layer: out[t,b] = g2[t,b,:] . Wt3[t,:,0] + bt3[t]
// grid (B/256, T)
// ---------------------------------------------------------------------------
__global__ __launch_bounds__(256) void tower3_kernel(
    const _Float16* __restrict__ g2, const float* __restrict__ Wt3,
    const float* __restrict__ bt3, float* __restrict__ out)
{
    __shared__ float w[TH2c];
    const int t = blockIdx.y;
    for (int i = threadIdx.x; i < TH2c; i += 256) w[i] = Wt3[(long long)t * TH2c + i];
    __syncthreads();
    const int b = blockIdx.x * 256 + threadIdx.x;
    const _Float16* g = g2 + ((long long)t * Bsz + b) * TH2c;
    float acc = bt3[t];
    for (int k = 0; k < TH2c; ++k) acc += (float)g[k] * w[k];
    out[(long long)t * Bsz + b] = acc;
}

// ---------------------------------------------------------------------------
extern "C" void kernel_launch(void* const* d_in, const int* in_sizes, int n_in,
                              void* d_out, int out_size, void* d_ws, size_t ws_size,
                              hipStream_t stream) {
    const float* x   = (const float*)d_in[0];
    const float* We1 = (const float*)d_in[1];
    const float* be1 = (const float*)d_in[2];
    const float* We2 = (const float*)d_in[3];
    const float* be2 = (const float*)d_in[4];
    const float* We3 = (const float*)d_in[5];
    const float* be3 = (const float*)d_in[6];
    const float* Wg  = (const float*)d_in[7];
    const float* Wt1 = (const float*)d_in[8];
    const float* bt1 = (const float*)d_in[9];
    const float* Wt2 = (const float*)d_in[10];
    const float* bt2 = (const float*)d_in[11];
    const float* Wt3 = (const float*)d_in[12];
    const float* bt3 = (const float*)d_in[13];
    float* out = (float*)d_out;

    // ---- workspace layout (bytes, 256-aligned) ----
    char* ws = (char*)d_ws;
    size_t o = 0;
    auto alloc = [&](size_t bytes) { size_t r = o; o += (bytes + 255) & ~(size_t)255; return r; };
    const size_t o_x16  = alloc((size_t)Bsz * Fdim * 2);
    const size_t o_We1f = alloc((size_t)Eexp * Fdim * EH1c * 2);
    const size_t o_We2f = alloc((size_t)Eexp * EH1c * EH2c * 2);
    const size_t o_We3f = alloc((size_t)Eexp * EH2c * Udim * 2);
    const size_t o_Wt1f = alloc((size_t)Tt * Udim * TH1c * 2);
    const size_t o_Wt2f = alloc((size_t)Tt * TH1c * TH2c * 2);
    const size_t o_bufA = alloc((size_t)Eexp * Bsz * EH1c * 2);  // h1, later eout
    const size_t o_bufB = alloc((size_t)Eexp * Bsz * EH2c * 2);  // h2, later gates/ti/g1/g2

    _Float16* x16  = (_Float16*)(ws + o_x16);
    _Float16* We1f = (_Float16*)(ws + o_We1f);
    _Float16* We2f = (_Float16*)(ws + o_We2f);
    _Float16* We3f = (_Float16*)(ws + o_We3f);
    _Float16* Wt1f = (_Float16*)(ws + o_Wt1f);
    _Float16* Wt2f = (_Float16*)(ws + o_Wt2f);
    _Float16* h1   = (_Float16*)(ws + o_bufA);
    _Float16* h2   = (_Float16*)(ws + o_bufB);
    _Float16* eout = (_Float16*)(ws + o_bufA);                 // alias (h1 dead)
    // sub-layout inside bufB after h2 is consumed:
    float*    gatesP = (float*)(ws + o_bufB);
    _Float16* ti16   = (_Float16*)(ws + o_bufB + (size_t)4 * Tt * Bsz * Eexp);
    _Float16* g1     = (_Float16*)((char*)ti16 + (size_t)Tt * Bsz * Udim * 2);
    _Float16* g2     = (_Float16*)((char*)g1 + (size_t)Tt * Bsz * TH1c * 2);

    // ---- fp32 -> f16 conversions ----
    cvt_f32_f16<<<4096, 256, 0, stream>>>(x,   x16,  (long long)Bsz * Fdim);
    cvt_f32_f16<<<4096, 256, 0, stream>>>(We1, We1f, (long long)Eexp * Fdim * EH1c);
    cvt_f32_f16<<<4096, 256, 0, stream>>>(We2, We2f, (long long)Eexp * EH1c * EH2c);
    cvt_f32_f16<<<4096, 256, 0, stream>>>(We3, We3f, (long long)Eexp * EH2c * Udim);
    cvt_f32_f16<<<2048, 256, 0, stream>>>(Wt1, Wt1f, (long long)Tt * Udim * TH1c);
    cvt_f32_f16<<<1024, 256, 0, stream>>>(Wt2, Wt2f, (long long)Tt * TH1c * TH2c);

    // ---- expert layer 1: h1[e] = relu(x @ We1[e] + be1[e])  (x stride 0) ----
    wmma_gemm_f16<<<dim3(EH1c / BN, Bsz / BM, Eexp), 256, 0, stream>>>(
        x16, 0LL, Fdim,
        We1f, (long long)Fdim * EH1c, EH1c,
        be1, EH1c,
        h1, (long long)Bsz * EH1c, EH1c,
        Bsz, EH1c, Fdim, 1);

    // ---- expert layer 2: h2[e] = relu(h1[e] @ We2[e] + be2[e]) ----
    wmma_gemm_f16<<<dim3(EH2c / BN, Bsz / BM, Eexp), 256, 0, stream>>>(
        h1, (long long)Bsz * EH1c, EH1c,
        We2f, (long long)EH1c * EH2c, EH2c,
        be2, EH2c,
        h2, (long long)Bsz * EH2c, EH2c,
        Bsz, EH2c, EH1c, 1);

    // ---- expert layer 3: eout[e] = h2[e] @ We3[e] + be3[e] (no relu) ----
    wmma_gemm_f16<<<dim3(Udim / BN, Bsz / BM, Eexp), 256, 0, stream>>>(
        h2, (long long)Bsz * EH2c, EH2c,
        We3f, (long long)EH2c * Udim, Udim,
        be3, Udim,
        eout, (long long)Bsz * Udim, Udim,
        Bsz, Udim, EH2c, 0);

    // ---- gates (h2 now dead; gates region aliases bufB) ----
    gates_kernel<<<dim3(Bsz / 256, Tt), 256, 0, stream>>>(x, Wg, gatesP);

    // ---- gate-weighted combine -> ti16 ----
    combine_kernel<<<(unsigned)((long long)Tt * Bsz * Udim / 256), 256, 0, stream>>>(
        gatesP, eout, ti16);

    // ---- tower layer 1: g1[t] = relu(ti[t] @ Wt1[t] + bt1[t]) ----
    wmma_gemm_f16<<<dim3(TH1c / BN, Bsz / BM, Tt), 256, 0, stream>>>(
        ti16, (long long)Bsz * Udim, Udim,
        Wt1f, (long long)Udim * TH1c, TH1c,
        bt1, TH1c,
        g1, (long long)Bsz * TH1c, TH1c,
        Bsz, TH1c, Udim, 1);

    // ---- tower layer 2: g2[t] = relu(g1[t] @ Wt2[t] + bt2[t]) ----
    wmma_gemm_f16<<<dim3(TH2c / BN, Bsz / BM, Tt), 256, 0, stream>>>(
        g1, (long long)Bsz * TH1c, TH1c,
        Wt2f, (long long)TH1c * TH2c, TH2c,
        bt2, TH2c,
        g2, (long long)Bsz * TH2c, TH2c,
        Bsz, TH2c, TH1c, 1);

    // ---- tower final layer -> out[t,b] ----
    tower3_kernel<<<dim3(Bsz / 256, Tt), 256, 0, stream>>>(g2, Wt3, bt3, out);
}
